// _DeformableCrossAttention_55619826483914
// MI455X (gfx1250) — matrix operand order repaired and checked by reading, not compiled, not run on hardware
//
#include <hip/hip_runtime.h>
#include <hip/hip_bf16.h>

typedef __attribute__((ext_vector_type(16))) __bf16 v16bf;
typedef __attribute__((ext_vector_type(2)))  __bf16 v2bf;
typedef __attribute__((ext_vector_type(8)))  float  v8f;

#define DM  256   // d_model
#define NHD 8     // heads
#define NPT 4     // points
#define HDm 32    // head dim
#define IH  128
#define IW  128
#define NQ  4096

// ---------------------------------------------------------------------------
// Weight prep: f32 -> bf16, pre-swizzled into WMMA B-fragment order:
//   Wsw[((s*NT + ntile)*32 + lane)*16 + i] = W[(s*32 + (lane>=16)*16 + i)][ntile*16 + (lane&15)]
// so each lane's 16 bf16 B-operand values are 32 contiguous bytes.
// Also fuses W_off||W_attn into one (256 x 96) matrix + fused bias.
// ---------------------------------------------------------------------------
__global__ __launch_bounds__(256) void prep_weights(
    const float* __restrict__ W_val, const float* __restrict__ W_out,
    const float* __restrict__ W_off, const float* __restrict__ b_off,
    const float* __restrict__ W_attn, const float* __restrict__ b_attn,
    __bf16* __restrict__ Wsw_val, __bf16* __restrict__ Wsw_out,
    __bf16* __restrict__ Wsw_qp, float* __restrict__ bqp) {
  const int t = blockIdx.x * 256 + threadIdx.x;  // 0..65535 -> (k, c)
  const int k = t >> 8;
  const int c = t & 255;
  const int s    = k >> 5;                  // K-step (32 wide)
  const int kk   = k & 31;                  // K within step
  const int lane = (c & 15) + ((kk >> 4) << 4);
  const int i    = kk & 15;

  const size_t d256 = ((size_t)(s * 16 + (c >> 4)) * 32 + lane) * 16 + i;
  Wsw_val[d256] = (__bf16)W_val[k * 256 + c];
  Wsw_out[d256] = (__bf16)W_out[k * 256 + c];
  if (c < 96) {
    const float w = (c < 64) ? W_off[k * 64 + c] : W_attn[k * 32 + (c - 64)];
    Wsw_qp[((size_t)(s * 6 + (c >> 4)) * 32 + lane) * 16 + i] = (__bf16)w;
    if (k == 0) bqp[c] = (c < 64) ? b_off[c] : b_attn[c];
  }
}

// ---------------------------------------------------------------------------
// Tiled GEMM: C[M,N] = A[M,256] (f32 -> bf16) @ W[256,N] (bf16, pre-swizzled)
//             + bias.  Block = 256 threads = 8 waves; each wave owns a 16-row
// M-slice and the full N width. LDS holds both operands in WMMA fragment
// order so every fragment load is one contiguous 32B LDS read.
// ---------------------------------------------------------------------------
template <int NT>
__global__ __launch_bounds__(256) void gemm_bf16_wmma(
    const float* __restrict__ A, const __bf16* __restrict__ Wsw,
    const float* __restrict__ bias, float* __restrict__ C, int M) {
  constexpr int N = NT * 16;
  __shared__ __bf16 sAf[8 * 32 * 16];    // 8 waves x 32 lanes x 16 (8 KB)
  __shared__ __bf16 sBf[NT * 32 * 16];   // NT tiles x 32 lanes x 16 (NT KB)

  const int tid  = threadIdx.x;
  const int wave = tid >> 5;
  const int lane = tid & 31;
  const int m0   = blockIdx.x * 128;
  const int hi   = (lane >= 16) ? 1 : 0;
  const int lr   = lane & 15;

  v8f acc[NT] = {};

  for (int s = 0; s < 8; ++s) {
    const int k0 = s * 32;

    // --- stage A (128x32 f32 -> bf16) into fragment layout ---
    // Each thread: 8 coalesced float2 loads; a pair never crosses an
    // 8-element K-run, so it lands as one aligned 4B LDS store.
#pragma unroll
    for (int it = 0; it < 8; ++it) {
      const int j = tid + it * 256;        // pair index 0..2047
      const int r = j >> 4;                // row 0..127
      const int c = (j & 15) * 2;          // even col 0..30
      const float2 a2 =
          *(const float2*)&A[(size_t)(m0 + r) * 256 + (k0 + c)];
      const int w  = r >> 4;
      const int ln = (r & 15) + ((c >> 4) << 4);
      const int i0 = c & 15;
      *(v2bf*)&sAf[(w * 32 + ln) * 16 + i0] = (v2bf){(__bf16)a2.x, (__bf16)a2.y};
    }

    // --- stage B chunk: straight coalesced copy (already swizzled) ---
    {
      const uint2* __restrict__ src = (const uint2*)(Wsw + (size_t)s * NT * 512);
      uint2* dst = (uint2*)sBf;
#pragma unroll
      for (int it = 0; it < NT / 2; ++it)
        dst[tid + it * 256] = src[tid + it * 256];
    }
    if (s < 7)  // prefetch next A tile (global_prefetch_b8)
      __builtin_prefetch(&A[(size_t)(m0 + (tid >> 1)) * 256 + (k0 + 32)], 0, 0);
    __syncthreads();

    // --- compute: one 32B LDS read per fragment ---
    const v16bf af = *(const v16bf*)&sAf[(wave * 32 + lane) * 16];
#pragma unroll
    for (int n = 0; n < NT; ++n) {
      const v16bf bfr = *(const v16bf*)&sBf[(n * 32 + lane) * 16];
      acc[n] = __builtin_amdgcn_wmma_f32_16x16x32_bf16(
          /*neg_a=*/false, af, /*neg_b=*/false, bfr,
          /*c_mod=*/(short)0, acc[n], /*reuse_a=*/false, /*reuse_b=*/false);
    }
    __syncthreads();
  }

  // --- store: VGPR r of lanes 0-15 -> M=r, lanes 16-31 -> M=8+r ---
#pragma unroll
  for (int n = 0; n < NT; ++n) {
    const int col = n * 16 + lr;
    const float bv = bias[col];
#pragma unroll
    for (int r = 0; r < 8; ++r) {
      const int row = m0 + wave * 16 + hi * 8 + r;
      C[(size_t)row * N + col] = acc[n][r] + bv;
    }
  }
}

// ---------------------------------------------------------------------------
// Deformable sampling: one wave per (b, q, h); lane = channel (HD=32).
// val_t layout (B, HW, NH, HD) keeps the 32 channels contiguous -> each
// corner gather is one coalesced 128B read that hits L2 (134MB < 192MB L2).
// ---------------------------------------------------------------------------
__global__ __launch_bounds__(256) void deform_sample(
    const float* __restrict__ val_t,   // (B, H*W, 256)
    const float* __restrict__ qp,      // (B*Nq, 96): [0..63]=offs, [64..95]=logits
    const float* __restrict__ refp,    // (B*Nq, 2)
    float* __restrict__ mid) {         // (B*Nq, 256)
  const int lane = threadIdx.x & 31;
  const int wid  = blockIdx.x * 8 + (threadIdx.x >> 5);
  const int h  = wid & (NHD - 1);
  const int bq = wid >> 3;
  const int b  = bq >> 12;             // Nq = 4096

  const float rx = refp[bq * 2 + 0];
  const float ry = refp[bq * 2 + 1];
  const float* q96 = qp + (size_t)bq * 96;

  // softmax over NP=4 logits for this head
  const float l0 = q96[64 + h * 4 + 0];
  const float l1 = q96[64 + h * 4 + 1];
  const float l2 = q96[64 + h * 4 + 2];
  const float l3 = q96[64 + h * 4 + 3];
  const float mx = fmaxf(fmaxf(l0, l1), fmaxf(l2, l3));
  const float e0 = __expf(l0 - mx), e1 = __expf(l1 - mx);
  const float e2 = __expf(l2 - mx), e3 = __expf(l3 - mx);
  const float inv = 1.0f / (e0 + e1 + e2 + e3);
  const float wts[4] = {e0 * inv, e1 * inv, e2 * inv, e3 * inv};

  const float* vb = val_t + (size_t)b * (IH * IW) * DM + h * HDm + lane;
  float acc = 0.0f;

#pragma unroll
  for (int p = 0; p < NPT; ++p) {
    const float ox = q96[h * 8 + p * 2 + 0];
    const float oy = q96[h * 8 + p * 2 + 1];
    // grid_sample (align_corners=False): x = px*W - 0.5
    const float x = (rx + ox) * (float)IW - 0.5f;
    const float y = (ry + oy) * (float)IH - 0.5f;
    const float x0f = floorf(x), y0f = floorf(y);
    const float wx1 = x - x0f, wy1 = y - y0f;
    const float wx0 = 1.0f - wx1, wy0 = 1.0f - wy1;
    const int x0 = (int)x0f, y0 = (int)y0f;
    const float wp = wts[p];

#pragma unroll
    for (int cy = 0; cy < 2; ++cy) {
#pragma unroll
      for (int cx = 0; cx < 2; ++cx) {
        const int xi = x0 + cx, yi = y0 + cy;
        const bool valid = ((unsigned)xi < (unsigned)IW) &&
                           ((unsigned)yi < (unsigned)IH);
        const float cw = (cx ? wx1 : wx0) * (cy ? wy1 : wy0);
        const int xc = min(max(xi, 0), IW - 1);
        const int yc = min(max(yi, 0), IH - 1);
        const float g = vb[(size_t)(yc * IW + xc) * DM];
        acc += g * (valid ? wp * cw : 0.0f);
      }
    }
  }
  mid[(size_t)bq * DM + h * HDm + lane] = acc;
}

// ---------------------------------------------------------------------------
extern "C" void kernel_launch(void* const* d_in, const int* in_sizes, int n_in,
                              void* d_out, int out_size, void* d_ws, size_t ws_size,
                              hipStream_t stream) {
  const float* query  = (const float*)d_in[0];
  const float* refp   = (const float*)d_in[1];
  const float* value  = (const float*)d_in[2];
  const float* W_off  = (const float*)d_in[3];
  const float* b_off  = (const float*)d_in[4];
  const float* W_attn = (const float*)d_in[5];
  const float* b_attn = (const float*)d_in[6];
  const float* W_val  = (const float*)d_in[7];
  const float* b_val  = (const float*)d_in[8];
  const float* W_out  = (const float*)d_in[9];
  const float* b_out  = (const float*)d_in[10];
  float* out = (float*)d_out;

  const int BNq = in_sizes[0] / DM;   // 32768
  const int BHW = in_sizes[2] / DM;   // 131072

  // workspace carve-out (256B aligned)
  char* ws = (char*)d_ws;
  size_t off = 0;
  auto carve = [&](size_t bytes) -> void* {
    void* p = ws + off;
    off = (off + bytes + 255) & ~(size_t)255;
    return p;
  };
  __bf16* Wsw_val = (__bf16*)carve((size_t)DM * DM * sizeof(__bf16));
  __bf16* Wsw_out = (__bf16*)carve((size_t)DM * DM * sizeof(__bf16));
  __bf16* Wsw_qp  = (__bf16*)carve((size_t)DM * 96 * sizeof(__bf16));
  float*  bqp     = (float*) carve(96 * sizeof(float));
  float*  val_t   = (float*) carve((size_t)BHW * DM * sizeof(float));   // 134 MB
  float*  qpbuf   = (float*) carve((size_t)BNq * 96 * sizeof(float));   // 12.6 MB
  float*  midbuf  = (float*) carve((size_t)BNq * DM * sizeof(float));   // 33.5 MB

  // 1) weights -> bf16, fragment-swizzled (+ fused qp weight/bias)
  prep_weights<<<256, 256, 0, stream>>>(W_val, W_out, W_off, b_off, W_attn,
                                        b_attn, Wsw_val, Wsw_out, Wsw_qp, bqp);

  // 2) value projection: (BHW x 256) @ (256 x 256) + b_val -> val_t (B,HW,NH,HD)
  gemm_bf16_wmma<16><<<BHW / 128, 256, 0, stream>>>(value, Wsw_val, b_val,
                                                    val_t, BHW);

  // 3) fused query projection: (BNq x 256) @ (256 x 96) + bqp -> qpbuf
  gemm_bf16_wmma<6><<<BNq / 128, 256, 0, stream>>>(query, Wsw_qp, bqp,
                                                   qpbuf, BNq);

  // 4) softmax + bilinear sampling + point-weighted sum -> midbuf (B,Nq,256)
  deform_sample<<<(BNq * NHD) / 8, 256, 0, stream>>>(val_t, qpbuf, refp,
                                                     midbuf);

  // 5) output projection: (BNq x 256) @ (256 x 256) + b_out -> d_out
  gemm_bf16_wmma<16><<<BNq / 128, 256, 0, stream>>>(midbuf, Wsw_out, b_out,
                                                    out, BNq);
}
